// UniPhyTransformerBlock_46231027974351
// MI455X (gfx1250) — matrix operand order repaired and checked by reading, not compiled, not run on hardware
//
#include <hip/hip_runtime.h>
#include <hip/hip_bf16.h>
#include <math.h>

#define BB 4
#define TT 16
#define HH 32
#define WW 32
#define CC 64
#define DD 128
#define EE 256
#define BT 64      /* BB*TT */
#define HW 1024    /* HH*WW */
#define NN 65536   /* BT*HW */
#define WF 17
#define RR 32
#define PW 34      /* padded spatial dim */
#define EPSLN 1e-5f

typedef __attribute__((ext_vector_type(16))) _Float16 v16h;
typedef __attribute__((ext_vector_type(8)))  _Float16 h8;
typedef __attribute__((ext_vector_type(8)))  float    v8f;

// ------------- WMMA fragment helpers (layouts per CDNA5 ISA) ------------
// A 16x32 f16: lane m=lane&15, g=lane>>4; element j -> k = j + 8*(g + (j>=8)).
// Contiguous as two 8-half chunks at k = 8g and k = 8g+16 -> two b128 loads.
__device__ __forceinline__ v16h ldAv(const _Float16* p, int stride, int kk){
  int lane = threadIdx.x & 31;
  int g = lane >> 4, m = lane & 15;
  const _Float16* q = p + (long)m*stride + kk + 8*g;
  h8 lo = *(const h8*)q;
  h8 hi = *(const h8*)(q + 16);
  return __builtin_shufflevector(lo, hi, 0,1,2,3,4,5,6,7,8,9,10,11,12,13,14,15);
}

// B 32x16 f16 from TRANSPOSED weights WT[n][k]: n=lane&15, k = kk + j + 8*(g + (j>=8))
// -> two 8-half chunks at k = 8g and k = 8g+16 -> two b128 loads.
__device__ __forceinline__ v16h ldBv(const _Float16* __restrict__ pT, int K, int col0, int kk){
  int lane = threadIdx.x & 31;
  int g = lane >> 4, n = lane & 15;
  const _Float16* q = pT + (long)(col0 + n)*K + kk + 8*g;
  h8 lo = *(const h8*)q;
  h8 hi = *(const h8*)(q + 16);
  return __builtin_shufflevector(lo, hi, 0,1,2,3,4,5,6,7,8,9,10,11,12,13,14,15);
}

__device__ __forceinline__ v8f wmma(v16h a, v16h b, v8f c){
  return __builtin_amdgcn_wmma_f32_16x16x32_f16(false, a, false, b, (short)0, c, false, false);
}

__device__ __forceinline__ float gelu_t(float x){
  float x3 = x*x*x;
  return 0.5f*x*(1.0f + tanhf(0.7978845608f*(x + 0.044715f*x3)));
}

// wave32 shfl reduction of two values + 4-partial LDS combine (block = 128 threads)
__device__ __forceinline__ void blockRed2(float& a, float& b, float* sh){
#pragma unroll
  for (int m = 16; m > 0; m >>= 1){
    a += __shfl_xor(a, m, 32);
    b += __shfl_xor(b, m, 32);
  }
  int lane = threadIdx.x & 31, wid = threadIdx.x >> 5;
  __syncthreads();
  if (lane == 0){ sh[2*wid] = a; sh[2*wid+1] = b; }
  __syncthreads();
  a = sh[0] + sh[2] + sh[4] + sh[6];
  b = sh[1] + sh[3] + sh[5] + sh[7];
}

// ------------- prep kernels --------------------------------------------------------
// conv weights (O,I,3,3) f32 -> TRANSPOSED f16 [tap][o][i] planes (wr, wi, -wi)
__global__ void k_prep_convw(const float* __restrict__ wr, const float* __restrict__ wi,
                             _Float16* __restrict__ pr, _Float16* __restrict__ pi,
                             _Float16* __restrict__ pin){
  int idx = blockIdx.x*blockDim.x + threadIdx.x;
  if (idx >= 9*CC*CC) return;
  int i   = idx % CC;
  int o   = (idx / CC) % CC;
  int tap = idx / (CC*CC);
  int in_idx = (o*CC + i)*9 + tap;
  float r = wr[in_idx], m = wi[in_idx];
  pr[idx]  = (_Float16)r;
  pi[idx]  = (_Float16)m;
  pin[idx] = (_Float16)(-m);
}

// dense weights [K][Ncol] f32 -> TRANSPOSED f16 [n][k] planes (wr, wi, -wi)
__global__ void k_prep_matT(const float* __restrict__ wr, const float* __restrict__ wi,
                            _Float16* __restrict__ pr, _Float16* __restrict__ pi,
                            _Float16* __restrict__ pin, int K, int Ncol){
  int idx = blockIdx.x*blockDim.x + threadIdx.x;
  if (idx >= K*Ncol) return;
  int k = idx % K;
  int n = idx / K;
  float r = wr[k*Ncol + n], m = wi[k*Ncol + n];
  pr[idx]  = (_Float16)r;
  pi[idx]  = (_Float16)m;
  pin[idx] = (_Float16)(-m);
}

// K = (ur + i ui)(D,R) @ (vr + i vi)(R,WF)
__global__ void k_prep_K(const float* __restrict__ ur, const float* __restrict__ ui,
                         const float* __restrict__ vr, const float* __restrict__ vi,
                         float* __restrict__ kr, float* __restrict__ ki){
  int idx = blockIdx.x*blockDim.x + threadIdx.x;
  if (idx >= DD*WF) return;
  int f = idx % WF, d = idx / WF;
  float ar = 0.f, ai = 0.f;
  for (int r = 0; r < RR; ++r){
    float a = ur[d*RR + r], b = ui[d*RR + r];
    float c = vr[r*WF + f], s = vi[r*WF + f];
    ar += a*c - b*s;
    ai += a*s + b*c;
  }
  kr[idx] = ar; ki[idx] = ai;
}

// zero-fill (float4 chunks)
__global__ void k_fillz(float4* __restrict__ p, int n){
  int i = blockIdx.x*blockDim.x + threadIdx.x;
  if (i < n) p[i] = make_float4(0.f, 0.f, 0.f, 0.f);
}

// ------------- stage 1: LN(time) + Hamiltonian scan --------------------------------
__global__ void k_scan(const float* __restrict__ x, const float* __restrict__ dt,
                       const float* __restrict__ g, const float* __restrict__ bta,
                       const float* __restrict__ om, const float* __restrict__ bin,
                       const float* __restrict__ cr, const float* __restrict__ ci,
                       const float* __restrict__ dcf,
                       float* __restrict__ out1, float* __restrict__ hout){
  __shared__ float sred[8];
  int site = blockIdx.x;            // b*HW + hw
  int b = site / HW, hw = site % HW;
  int d = threadIdx.x;
  float gd = g[d], bd = bta[d];
  float omd = om[d], bind = bin[d], crd = cr[d], cid = ci[d], ddc = dcf[d];
  float hr = 0.f, hi = 0.f;
  for (int t = 0; t < TT; ++t){
    long idx = (((long)(b*TT + t))*HW + hw)*DD + d;
    float v = x[idx];
    float sa = v, sb = v*v;
    blockRed2(sa, sb, sred);
    float mu  = sa*(1.0f/DD);
    float var = sb*(1.0f/DD) - mu*mu;
    float xn = (v - mu)*rsqrtf(var + EPSLN)*gd + bd;
    float dtt = dt[b*TT + t];
    float sph, cph; sincosf(omd*dtt, &sph, &cph);
    float hrn = cph*hr - sph*hi + bind*xn;
    float hin = sph*hr + cph*hi;
    hr = hrn; hi = hin;
    float y = crd*hr - cid*hi + ddc*xn;
    out1[idx] = y + v;               // + residual (original x)
  }
  long ho = (((long)site)*DD + d)*2;
  hout[ho]   = hr;
  hout[ho+1] = hi;
}

// ------------- stage 2a: LN(space) -> f32 planes + padded channel-last f16 ---------
__global__ void k_lnspace(const float* __restrict__ in, const float* __restrict__ g,
                          const float* __restrict__ bta, float* __restrict__ P,
                          _Float16* __restrict__ Ar, _Float16* __restrict__ Ai){
  __shared__ float sred[8];
  int n = blockIdx.x;
  int d = threadIdx.x;
  float v = in[(long)n*DD + d];
  float sa = v, sb = v*v;
  blockRed2(sa, sb, sred);
  float mu  = sa*(1.f/DD);
  float var = sb*(1.f/DD) - mu*mu;
  float xn = (v - mu)*rsqrtf(var + EPSLN)*g[d] + bta[d];
  int bt = n / HW, hw = n % HW;
  int h = hw >> 5, w = hw & 31;
  P[((long)bt*DD + d)*HW + hw] = xn;
  int c = d >> 1;
  long pidx = (((long)bt*PW + (h+1))*PW + (w+1))*CC + c;
  if ((d & 1) == 0) Ar[pidx] = (_Float16)xn;
  else              Ai[pidx] = (_Float16)xn;
}

// ------------- stage 2b: complex 3x3 conv, implicit-GEMM WMMA with LDS tile --------
// block = 128 (4 waves), waves share the LDS A-tile, each owns one 16-outch tile.
__global__ void k_conv(const _Float16* __restrict__ Ar, const _Float16* __restrict__ Ai,
                       const _Float16* __restrict__ WrT, const _Float16* __restrict__ WiT,
                       const _Float16* __restrict__ WinT,
                       const float* __restrict__ br, const float* __restrict__ bi,
                       float* __restrict__ S){
  __shared__ __align__(16) _Float16 sAr[3*18*CC];
  __shared__ __align__(16) _Float16 sAi[3*18*CC];
  int tile = blockIdx.x;            // bt*64 + h*2 + wseg
  int wseg = tile & 1;
  int h = (tile >> 1) & 31;
  int bt = tile >> 6;
  int w0 = wseg * 16;
  int tid = threadIdx.x;

  // stage padded region rows h..h+2, cols w0..w0+17 (channel-last) into LDS, b128s
  const int CHUNKS = 3*18*CC/8;     // 432 per plane
  for (int i = tid; i < 2*CHUNKS; i += 128){
    int plane = (i >= CHUNKS);
    int c8 = plane ? i - CHUNKS : i;
    int seg = c8 & 7;               // 8 chunks of 8 halves per pixel
    int pix = c8 >> 3;              // 0..53
    int r3 = pix / 18, c18 = pix % 18;
    long gidx = (((long)bt*PW + (h + r3))*PW + (w0 + c18))*CC + seg*8;
    const h8* src = (const h8*)((plane ? Ai : Ar) + gidx);
    h8* dst = (h8*)((plane ? sAi : sAr) + pix*CC) + seg;
    *dst = *src;
  }
  __syncthreads();

  int wid = tid >> 5, lane = tid & 31;
  int g = lane >> 4, nc = lane & 15;
  int obase = wid * 16;
  v8f accr = {0.f,0.f,0.f,0.f,0.f,0.f,0.f,0.f};
  v8f acci = {0.f,0.f,0.f,0.f,0.f,0.f,0.f,0.f};
  for (int tap = 0; tap < 9; ++tap){
    int dy = tap/3, dx = tap%3;     // padded offsets 0..2
    const _Float16* pr = sAr + (dy*18 + dx)*CC;
    const _Float16* pi = sAi + (dy*18 + dx)*CC;
    const _Float16* wrt = WrT  + tap*CC*CC;
    const _Float16* wit = WiT  + tap*CC*CC;
    const _Float16* wnt = WinT + tap*CC*CC;
#pragma unroll
    for (int kk = 0; kk < CC; kk += 32){
      v16h ar = ldAv(pr, CC, kk);
      v16h ai = ldAv(pi, CC, kk);
      v16h bwr  = ldBv(wrt, CC, obase, kk);
      v16h bwi  = ldBv(wit, CC, obase, kk);
      v16h bwin = ldBv(wnt, CC, obase, kk);
      accr = wmma(ar, bwr,  accr);   // a*wr
      accr = wmma(ai, bwin, accr);   // - b*wi
      acci = wmma(ar, bwi,  acci);   // a*wi
      acci = wmma(ai, bwr,  acci);   // + b*wr
    }
  }
#pragma unroll
  for (int r = 0; r < 8; ++r){
    int mm = r + 8*g;
    int oc = obase + nc;
    long sidx = (((long)bt*DD + 2*oc)*HW) + (long)h*WW + w0 + mm;
    S[sidx]      = accr[r] + br[oc];   // d = 2c   (real)
    S[sidx + HW] = acci[r] + bi[oc];   // d = 2c+1 (imag)
  }
}

// ------------- stage 2c: spectral mix along W (rfft*K*mod -> irfft), S += ----------
__global__ void k_spec(const float* __restrict__ P, const float* __restrict__ Kr,
                       const float* __restrict__ Ki, const float* __restrict__ dt,
                       const float* __restrict__ sdec, const float* __restrict__ som,
                       float* __restrict__ S){
  __shared__ float xrow[8][WW];
  __shared__ float Yr[8][WF], Yi[8][WF];
  __shared__ float ct[WW], st[WW];
  int tid = threadIdx.x;
  if (tid < WW){
    float ang = 6.2831853071795864f * (float)tid * (1.0f/WW);
    float s, c; sincosf(ang, &s, &c);
    ct[tid] = c; st[tid] = s;
  }
  int rid  = tid >> 5;
  int lane = tid & 31;
  long row = (long)blockIdx.x*8 + rid;   // (bt*DD + d)*HH + h
  int h = (int)(row % HH);
  long btd = row / HH;
  int d  = (int)(btd % DD);
  int bt = (int)(btd / DD);
  long base = btd*HW + (long)h*WW;
  xrow[rid][lane] = P[base + lane];
  __syncthreads();
  if (lane < WF){
    float sr = 0.f, si = 0.f;
#pragma unroll
    for (int w = 0; w < WW; ++w){
      int k = (lane*w) & 31;
      float xv = xrow[rid][w];
      sr += xv*ct[k];
      si -= xv*st[k];                 // forward DFT: e^{-i2pi fw/32}
    }
    float kr = Kr[d*WF + lane], ki = Ki[d*WF + lane];
    float tr = sr*kr - si*ki;
    float ti = sr*ki + si*kr;
    float dec = sdec[lane];
    float sp = (dec > 20.f) ? dec : log1pf(expf(dec));   // softplus
    float dtb = dt[bt];
    float amp = expf(-sp*dtb);
    float ms, mc; sincosf(som[lane]*dtb, &ms, &mc);
    Yr[rid][lane] = tr*(amp*mc) - ti*(amp*ms);
    Yi[rid][lane] = tr*(amp*ms) + ti*(amp*mc);
  }
  __syncthreads();
  float acc = 0.f;
#pragma unroll
  for (int f = 0; f < WF; ++f){
    float coef = (f == 0 || f == WF-1) ? 1.0f : 2.0f;
    int k = (f*lane) & 31;
    acc += coef*(Yr[rid][f]*ct[k] - Yi[rid][f]*st[k]);
  }
  S[base + lane] += acc * (1.0f/WW);
}

// ------------- stage 3a: residual + LN(feat) -> f16 X rows -------------------------
__global__ void k_fuse(const float* __restrict__ S, const float* __restrict__ res2,
                       const float* __restrict__ g, const float* __restrict__ bta,
                       float* __restrict__ res3, _Float16* __restrict__ Xr,
                       _Float16* __restrict__ Xi){
  __shared__ float sred[8];
  int n = blockIdx.x;
  int d = threadIdx.x;
  int bt = n / HW, hw = n % HW;
  float v = S[((long)bt*DD + d)*HW + hw] + res2[(long)n*DD + d];
  res3[(long)n*DD + d] = v;
  float sa = v, sb = v*v;
  blockRed2(sa, sb, sred);
  float mu  = sa*(1.f/DD);
  float var = sb*(1.f/DD) - mu*mu;
  float xn = (v - mu)*rsqrtf(var + EPSLN)*g[d] + bta[d];
  int c = d >> 1;
  if ((d & 1) == 0) Xr[(long)n*CC + c] = (_Float16)xn;
  else              Xi[(long)n*CC + c] = (_Float16)xn;
}

// ------------- stage 3b: complex GEMM1 (64->256) + gelu, LDS-staged A --------------
__global__ void k_gemm1(const _Float16* __restrict__ Xr, const _Float16* __restrict__ Xi,
                        const _Float16* __restrict__ WrT, const _Float16* __restrict__ WiT,
                        const _Float16* __restrict__ WinT,
                        const float* __restrict__ b1r, const float* __restrict__ b1i,
                        _Float16* __restrict__ Hr, _Float16* __restrict__ Hi){
  __shared__ __align__(16) _Float16 sXr[16*CC], sXi[16*CC];
  int row0 = blockIdx.x * 16;
  int tid = threadIdx.x;
  {
    int plane = tid >> 7;           // 256 threads, 128 b128-chunks per plane
    int c8 = tid & 127;
    const h8* src = (const h8*)((plane ? Xi : Xr) + (long)row0*CC) + c8;
    ((h8*)(plane ? sXi : sXr))[c8] = *src;
  }
  __syncthreads();
  int wid  = tid >> 5;
  int lane = tid & 31;
  int col0 = (blockIdx.y*8 + wid) * 16;
  v8f accr = {0.f,0.f,0.f,0.f,0.f,0.f,0.f,0.f};
  v8f acci = {0.f,0.f,0.f,0.f,0.f,0.f,0.f,0.f};
#pragma unroll
  for (int kk = 0; kk < CC; kk += 32){
    v16h axr = ldAv(sXr, CC, kk);
    v16h axi = ldAv(sXi, CC, kk);
    v16h bwr  = ldBv(WrT,  CC, col0, kk);
    v16h bwi  = ldBv(WiT,  CC, col0, kk);
    v16h bwin = ldBv(WinT, CC, col0, kk);
    accr = wmma(axr, bwr,  accr);
    accr = wmma(axi, bwin, accr);
    acci = wmma(axr, bwi,  acci);
    acci = wmma(axi, bwr,  acci);
  }
  int g = lane >> 4, nn = lane & 15;
#pragma unroll
  for (int r = 0; r < 8; ++r){
    int m = r + 8*g;
    int col = col0 + nn;
    float hrv = gelu_t(accr[r] + b1r[col]);
    float hiv = gelu_t(acci[r] + b1i[col]);
    Hr[(long)(row0+m)*EE + col] = (_Float16)hrv;
    Hi[(long)(row0+m)*EE + col] = (_Float16)hiv;
  }
}

// ------------- stage 3c: complex GEMM2 (256->64) + bias + residual -----------------
__global__ void k_gemm2(const _Float16* __restrict__ Hr, const _Float16* __restrict__ Hi,
                        const _Float16* __restrict__ WrT, const _Float16* __restrict__ WiT,
                        const _Float16* __restrict__ WinT,
                        const float* __restrict__ b2r, const float* __restrict__ b2i,
                        const float* __restrict__ res3, float* __restrict__ xout){
  __shared__ __align__(16) _Float16 sHr[16*EE], sHi[16*EE];
  int row0 = blockIdx.x * 16;
  int tid = threadIdx.x;
  for (int i = tid; i < 1024; i += 128){   // 512 b128-chunks per plane
    int plane = (i >= 512);
    int c8 = plane ? i - 512 : i;
    const h8* src = (const h8*)((plane ? Hi : Hr) + (long)row0*EE) + c8;
    ((h8*)(plane ? sHi : sHr))[c8] = *src;
  }
  __syncthreads();
  int wid  = tid >> 5;
  int lane = tid & 31;
  int col0 = wid * 16;
  v8f accr = {0.f,0.f,0.f,0.f,0.f,0.f,0.f,0.f};
  v8f acci = {0.f,0.f,0.f,0.f,0.f,0.f,0.f,0.f};
#pragma unroll
  for (int kk = 0; kk < EE; kk += 32){
    if (kk + 32 < EE){
      __builtin_prefetch((const void*)(WrT + (long)(col0 + lane)*EE + kk + 32), 0, 1);
      __builtin_prefetch((const void*)(WiT + (long)(col0 + lane)*EE + kk + 32), 0, 1);
    }
    v16h ahr = ldAv(sHr, EE, kk);
    v16h ahi = ldAv(sHi, EE, kk);
    v16h bwr  = ldBv(WrT,  EE, col0, kk);
    v16h bwi  = ldBv(WiT,  EE, col0, kk);
    v16h bwin = ldBv(WinT, EE, col0, kk);
    accr = wmma(ahr, bwr,  accr);
    accr = wmma(ahi, bwin, accr);
    acci = wmma(ahr, bwi,  acci);
    acci = wmma(ahi, bwr,  acci);
  }
  int g = lane >> 4, nn = lane & 15;
#pragma unroll
  for (int r = 0; r < 8; ++r){
    int m = r + 8*g;
    int col = col0 + nn;
    long rowi = row0 + m;
    long od = (rowi*CC + col)*2;
    xout[od]   = accr[r] + b2r[col] + res3[rowi*DD + 2*col];
    xout[od+1] = acci[r] + b2i[col] + res3[rowi*DD + 2*col + 1];
  }
}

// ------------- host launcher -------------------------------------------------------
extern "C" void kernel_launch(void* const* d_in, const int* in_sizes, int n_in,
                              void* d_out, int out_size, void* d_ws, size_t ws_size,
                              hipStream_t stream) {
  const float* x        = (const float*)d_in[0];
  const float* dt       = (const float*)d_in[1];
  const float* ln_t_g   = (const float*)d_in[2];
  const float* ln_t_b   = (const float*)d_in[3];
  const float* ln_s_g   = (const float*)d_in[4];
  const float* ln_s_b   = (const float*)d_in[5];
  const float* ln_f_g   = (const float*)d_in[6];
  const float* ln_f_b   = (const float*)d_in[7];
  const float* ham_om   = (const float*)d_in[8];
  const float* ham_bin  = (const float*)d_in[9];
  const float* ham_cr   = (const float*)d_in[10];
  const float* ham_ci   = (const float*)d_in[11];
  const float* ham_d    = (const float*)d_in[12];
  const float* cliff_wr = (const float*)d_in[13];
  const float* cliff_wi = (const float*)d_in[14];
  const float* cliff_br = (const float*)d_in[15];
  const float* cliff_bi = (const float*)d_in[16];
  const float* spec_ur  = (const float*)d_in[17];
  const float* spec_ui  = (const float*)d_in[18];
  const float* spec_vr  = (const float*)d_in[19];
  const float* spec_vi  = (const float*)d_in[20];
  const float* spec_dec = (const float*)d_in[21];
  const float* spec_om  = (const float*)d_in[22];
  const float* w1r      = (const float*)d_in[23];
  const float* w1i      = (const float*)d_in[24];
  const float* b1r      = (const float*)d_in[25];
  const float* b1i      = (const float*)d_in[26];
  const float* w2r      = (const float*)d_in[27];
  const float* w2i      = (const float*)d_in[28];
  const float* b2r      = (const float*)d_in[29];
  const float* b2i      = (const float*)d_in[30];

  float* xout = (float*)d_out;                          // N*C*2 floats
  float* hout = xout + (size_t)NN*CC*2;                 // B*HW*D*2 floats

  const size_t PADE = (size_t)BT*PW*PW*CC;              // padded f16 plane elements

  char* wp = (char*)d_ws;
  auto alloc = [&](size_t bytes) -> void* {
    void* p = (void*)wp;
    wp += (bytes + 255) & ~(size_t)255;
    return p;
  };
  float*    out1 = (float*)   alloc((size_t)NN*DD*4);   // y + residual (res2)
  float*    Sbuf = (float*)   alloc((size_t)NN*DD*4);   // conv + spectral planes
  float*    Pbuf = (float*)   alloc((size_t)NN*DD*4);   // ln_space planes; reused as res3
  _Float16* Arb  = (_Float16*)alloc(PADE*2);            // padded ch-last; reused as MLP Xr
  _Float16* Aib  = (_Float16*)alloc(PADE*2);            // padded ch-last; reused as MLP Xi
  _Float16* Hrb  = (_Float16*)alloc((size_t)NN*EE*2);
  _Float16* Hib  = (_Float16*)alloc((size_t)NN*EE*2);
  _Float16* WcrT = (_Float16*)alloc((size_t)9*CC*CC*2);
  _Float16* WciT = (_Float16*)alloc((size_t)9*CC*CC*2);
  _Float16* WcnT = (_Float16*)alloc((size_t)9*CC*CC*2);
  _Float16* W1rT = (_Float16*)alloc((size_t)CC*EE*2);
  _Float16* W1iT = (_Float16*)alloc((size_t)CC*EE*2);
  _Float16* W1nT = (_Float16*)alloc((size_t)CC*EE*2);
  _Float16* W2rT = (_Float16*)alloc((size_t)EE*CC*2);
  _Float16* W2iT = (_Float16*)alloc((size_t)EE*CC*2);
  _Float16* W2nT = (_Float16*)alloc((size_t)EE*CC*2);
  float*    Kcr  = (float*)   alloc((size_t)DD*WF*4);
  float*    Kci  = (float*)   alloc((size_t)DD*WF*4);

  // prep: transposed f16 weights, spectral K, zero padded planes
  k_prep_convw<<<(9*CC*CC + 255)/256, 256, 0, stream>>>(cliff_wr, cliff_wi, WcrT, WciT, WcnT);
  k_prep_matT <<<(CC*EE + 255)/256,   256, 0, stream>>>(w1r, w1i, W1rT, W1iT, W1nT, CC, EE);
  k_prep_matT <<<(EE*CC + 255)/256,   256, 0, stream>>>(w2r, w2i, W2rT, W2iT, W2nT, EE, CC);
  k_prep_K    <<<(DD*WF + 255)/256,   256, 0, stream>>>(spec_ur, spec_ui, spec_vr, spec_vi, Kcr, Kci);
  {
    int nch = (int)(PADE/8);   // f16 elems per float4
    k_fillz<<<(nch + 255)/256, 256, 0, stream>>>((float4*)Arb, nch);
    k_fillz<<<(nch + 255)/256, 256, 0, stream>>>((float4*)Aib, nch);
  }

  // stage 1: LN(time) + scan + residual, h_final -> hout
  k_scan<<<BB*HW, DD, 0, stream>>>(x, dt, ln_t_g, ln_t_b, ham_om, ham_bin,
                                   ham_cr, ham_ci, ham_d, out1, hout);

  // stage 2: LN(space) -> planes; WMMA conv; spectral add
  k_lnspace<<<NN, DD, 0, stream>>>(out1, ln_s_g, ln_s_b, Pbuf, Arb, Aib);
  k_conv<<<BT*HH*2, 128, 0, stream>>>(Arb, Aib, WcrT, WciT, WcnT, cliff_br, cliff_bi, Sbuf);
  k_spec<<<(BT*DD*HH)/8, 256, 0, stream>>>(Pbuf, Kcr, Kci, dt, spec_dec, spec_om, Sbuf);

  // stage 3: residual + LN(feat) -> X; WMMA MLP; final residual -> xout
  k_fuse<<<NN, DD, 0, stream>>>(Sbuf, out1, ln_f_g, ln_f_b, Pbuf, Arb, Aib);
  k_gemm1<<<dim3(NN/16, 2), 256, 0, stream>>>(Arb, Aib, W1rT, W1iT, W1nT, b1r, b1i, Hrb, Hib);
  k_gemm2<<<NN/16, 128, 0, stream>>>(Hrb, Hib, W2rT, W2iT, W2nT, b2r, b2i, Pbuf, xout);
}
